// Attention_82136954569171
// MI455X (gfx1250) — matrix operand order repaired and checked by reading, not compiled, not run on hardware
//
#include <hip/hip_runtime.h>
#include <hip/hip_bf16.h>

typedef _Float16 v16h __attribute__((ext_vector_type(16)));
typedef _Float16 v8h  __attribute__((ext_vector_type(8)));
typedef _Float16 v4h  __attribute__((ext_vector_type(4)));
typedef float    v8f  __attribute__((ext_vector_type(8)));
typedef float    v4f  __attribute__((ext_vector_type(4)));
typedef unsigned int u32x4 __attribute__((ext_vector_type(4)));
typedef int          i32x4 __attribute__((ext_vector_type(4)));
typedef int          i32x8 __attribute__((ext_vector_type(8)));

#define BATCH  4
#define SEQ    2048
#define NHEADS 12
#define DHEAD  64
#define DMODEL 768
#define ROWS   (BATCH*SEQ)   // 8192

#if defined(__has_builtin)
#if __has_builtin(__builtin_amdgcn_tensor_load_to_lds) && \
    __has_builtin(__builtin_amdgcn_s_wait_tensorcnt)
#define HAVE_TDM 1
#endif
#endif
#ifndef HAVE_TDM
#define HAVE_TDM 0
#endif

// ---- WMMA helper: D = A(16x32 f16) * B(32x16 f16) + C(16x16 f32) ----
__device__ __forceinline__ v8f wmma_f16(v16h a, v16h b, v8f c) {
  return __builtin_amdgcn_wmma_f32_16x16x32_f16(
      false, a, false, b, (short)0, c, false, false);
}

__device__ __forceinline__ v16h pack16(v8h lo, v8h hi) {
  v16h r;
#pragma unroll
  for (int i = 0; i < 8; i++) { r[i] = lo[i]; r[i + 8] = hi[i]; }
  return r;
}

// Low 32 bits of a flat shared-memory address are the LDS byte offset
// (aperture places the offset in addr[31:0], per CDNA5 ISA 10.2).
__device__ __forceinline__ unsigned lds_off(const void* p) {
  return (unsigned)(unsigned long long)(uintptr_t)p;
}

#if HAVE_TDM
// Issue a TDM DMA of a 2D f16 tile [tile1 rows x tile0 cols] starting at
// gtile (row stride = stride0 elements) into LDS at lds_addr (packed
// row-major [tile1][tile0]). Wave-level; caller guards to one wave.
__device__ __forceinline__ void tdm_load_tile_f16(
    const _Float16* gtile, unsigned lds_addr,
    unsigned td0, unsigned td1, unsigned stride0,
    unsigned tile0, unsigned tile1) {
  unsigned long long ga = (unsigned long long)(uintptr_t)gtile;
  u32x4 g0;
  g0[0] = 1u;                                            // count=1, load
  g0[1] = lds_addr;                                      // lds_addr[63:32]
  g0[2] = (unsigned)ga;                                  // global_addr lo
  g0[3] = ((unsigned)(ga >> 32) & 0x1FFFFFFu) | (2u << 30); // addr hi | type=2
  i32x8 g1;
  g1[0] = 0x10000;                                       // data_size=2B
  g1[1] = (int)(td0 << 16);                              // tensor_dim0 lo16
  g1[2] = (int)((td0 >> 16) | (td1 << 16));              // dim0 hi | dim1 lo
  g1[3] = (int)((td1 >> 16) | (tile0 << 16));            // dim1 hi | tile_dim0
  g1[4] = (int)tile1;                                    // tile_dim1 (dim2=0)
  g1[5] = (int)stride0;                                  // dim0_stride lo32
  g1[6] = 0;                                             // stride hi | d1stride
  g1[7] = 0;
  i32x4 z4; z4[0] = 0; z4[1] = 0; z4[2] = 0; z4[3] = 0;
#if __clang_major__ >= 23
  i32x8 z8;
#pragma unroll
  for (int i = 0; i < 8; i++) z8[i] = 0;
  __builtin_amdgcn_tensor_load_to_lds(g0, g1, z4, z4, z8, 0);
#else
  __builtin_amdgcn_tensor_load_to_lds(g0, g1, z4, z4, 0);
#endif
}
#endif

// ---------------------------------------------------------------------------
// Kernel 0a: convert x fp32 -> f16 once.
// ---------------------------------------------------------------------------
__global__ __launch_bounds__(256) void cvt_x_kernel(
    const float* __restrict__ x, _Float16* __restrict__ xh) {
  int idx = (blockIdx.x * 256 + threadIdx.x) * 4;
  v4f xv = *(const v4f*)(x + idx);
  v4h hv;
  hv[0] = (_Float16)xv[0]; hv[1] = (_Float16)xv[1];
  hv[2] = (_Float16)xv[2]; hv[3] = (_Float16)xv[3];
  *(v4h*)(xh + idx) = hv;
}

// ---------------------------------------------------------------------------
// Kernel 0b: pack weights to f16, contraction dim contiguous per row.
//   wqT/wkT/wvT : [nh=768][e=768],  woT : [e=768][nh=768]
// ---------------------------------------------------------------------------
__global__ __launch_bounds__(256) void pack_weights_kernel(
    const float* __restrict__ WQ, const float* __restrict__ WK,
    const float* __restrict__ WV, const float* __restrict__ WO,
    _Float16* __restrict__ wqT, _Float16* __restrict__ wkT,
    _Float16* __restrict__ wvT, _Float16* __restrict__ woT) {
  int idx = blockIdx.x * 256 + threadIdx.x;          // over [nh][e]
  if (idx >= DMODEL * DMODEL) return;
  int nh = idx / DMODEL;
  int e  = idx - nh * DMODEL;
  int n = nh >> 6, h = nh & 63;
  int src = (n * DMODEL + e) * DHEAD + h;            // W_{Q,K,V}[n][e][h]
  wqT[idx] = (_Float16)WQ[src];
  wkT[idx] = (_Float16)WK[src];
  wvT[idx] = (_Float16)WV[src];
  woT[e * DMODEL + nh] = (_Float16)WO[nh * DMODEL + e];  // W_O[n][h][e]
}

// ---------------------------------------------------------------------------
// Kernel 1: QKV projection via TDM-staged LDS tiles + WMMA.
// Block 256 thr (8 waves), output tile 128(M) x 64(N); head = blockIdx.y.
// ---------------------------------------------------------------------------
__global__ __launch_bounds__(256) void qkv_proj_kernel(
    const _Float16* __restrict__ xh,
    const _Float16* __restrict__ wqT, const _Float16* __restrict__ wkT,
    const _Float16* __restrict__ wvT,
    const float* __restrict__ bQ, const float* __restrict__ bK,
    const float* __restrict__ bV,
    _Float16* __restrict__ qh, _Float16* __restrict__ kh,
    _Float16* __restrict__ vT) {
  __shared__ alignas(16) _Float16 alds[2][128 * 32];   // 16 KB x-tiles
  __shared__ alignas(16) _Float16 wlds[2][64 * 32];    //  8 KB W-tiles

  const int which   = blockIdx.z;                      // 0=Q 1=K 2=V
  const int head    = blockIdx.y;
  const int rowbase = blockIdx.x * 128;
  const _Float16* W    = (which == 0) ? wqT : (which == 1) ? wkT : wvT;
  const float*    bias = (which == 0) ? bQ  : (which == 1) ? bK  : bV;
  const _Float16* xbase = xh + (size_t)rowbase * DMODEL;
  const _Float16* wbase = W + (size_t)(head * 64) * DMODEL;

  const int tid = threadIdx.x;
  const int wid = tid >> 5;
  const int L   = tid & 31;
  const int M   = L & 15;
  const int hs  = L >> 4;
  const int kbA = hs ? 8 : 0;       // A-layout lane-half K offset
  const int kbB = hs ? 16 : 0;      // unused; B fragments also load with kbA

  v8f acc[4];
#pragma unroll
  for (int t = 0; t < 4; t++)
#pragma unroll
    for (int v = 0; v < 8; v++) acc[t][v] = 0.f;

  auto issue = [&](int buf, int k0) {
#if HAVE_TDM
    if (wid == 0) {   // wave-uniform guard: TDM ignores EXEC, issue once
      tdm_load_tile_f16(xbase + k0, lds_off(&alds[buf][0]),
                        DMODEL, ROWS, DMODEL, 32, 128);
      tdm_load_tile_f16(wbase + k0, lds_off(&wlds[buf][0]),
                        DMODEL, DMODEL, DMODEL, 32, 64);
    }
#else
#pragma unroll
    for (int i = 0; i < 2; i++) {  // 128x32 f16 copy
      int idx = tid + 256 * i, row = idx >> 2, c8 = (idx & 3) * 8;
      *(v8h*)&alds[buf][row * 32 + c8] =
          *(const v8h*)(xbase + (size_t)row * DMODEL + k0 + c8);
    }
    { int row = tid >> 2, c8 = (tid & 3) * 8;  // 64x32 f16 copy
      *(v8h*)&wlds[buf][row * 32 + c8] =
          *(const v8h*)(wbase + (size_t)row * DMODEL + k0 + c8); }
#endif
  };
  auto settle = [&]() {
#if HAVE_TDM
    if (wid == 0) __builtin_amdgcn_s_wait_tensorcnt((short)0);
#endif
    __syncthreads();
  };

  issue(0, 0);
  settle();
#pragma unroll 1
  for (int ks = 0; ks < 24; ks++) {
    int cur = ks & 1;
    if (ks + 1 < 24) issue(cur ^ 1, (ks + 1) * 32);   // overlap DMA w/ WMMA
    // Batch all fragment loads first (clause + single dscnt wait), then WMMAs
    const _Float16* arow = &alds[cur][(wid * 16 + M) * 32];
    v16h af = pack16(*(const v8h*)(arow + kbA), *(const v8h*)(arow + 16 + kbA));
    v16h bf[4];
#pragma unroll
    for (int t = 0; t < 4; t++) {
      const _Float16* wrow = &wlds[cur][(t * 16 + M) * 32];
      bf[t] = pack16(*(const v8h*)(wrow + kbA), *(const v8h*)(wrow + kbA + 16));
    }
#pragma unroll
    for (int t = 0; t < 4; t++) acc[t] = wmma_f16(af, bf[t], acc[t]);
    settle();
  }

#pragma unroll
  for (int t = 0; t < 4; t++) {
    int c = t * 16 + M;
    float bb = bias[head * 64 + c];
#pragma unroll
    for (int v = 0; v < 8; v++) {
      float val = acc[t][v] + bb;
      int R = rowbase + wid * 16 + v + 8 * hs;
      int b = R >> 11, s = R & 2047;
      size_t hb = (size_t)(b * NHEADS + head);
      if (which == 0) {
        qh[((hb * SEQ + s) << 6) + c] = (_Float16)(val * 0.125f); // 1/sqrt(64)
      } else if (which == 1) {
        kh[((hb * SEQ + s) << 6) + c] = (_Float16)val;
      } else {
        vT[(hb * DHEAD + c) * SEQ + s] = (_Float16)val;  // V transposed [h][s]
      }
    }
  }
}

// ---------------------------------------------------------------------------
// Kernel 2: causal flash attention per (b, head). One wave per 16 query
// rows; key blocks of 32; scores & P*V via WMMA; online softmax via
// 16-lane shfl reductions; P C->A layout transpose via per-wave LDS.
// ---------------------------------------------------------------------------
__global__ __launch_bounds__(256) void attn_kernel(
    const _Float16* __restrict__ qh, const _Float16* __restrict__ kh,
    const _Float16* __restrict__ vT, _Float16* __restrict__ zb) {
  __shared__ alignas(16) _Float16 plds[8][16 * 32];

  const int b = blockIdx.z, n = blockIdx.y;
  const int wid = threadIdx.x >> 5, L = threadIdx.x & 31;
  const int M = L & 15, hs = L >> 4;
  const int kbA = hs ? 8 : 0, kbB = hs ? 16 : 0;

  const size_t hb = (size_t)(b * NHEADS + n);
  const _Float16* Q = qh + hb * SEQ * DHEAD;
  const _Float16* K = kh + hb * SEQ * DHEAD;
  const _Float16* V = vT + hb * DHEAD * SEQ;          // [h][s]
  const int q0 = (blockIdx.x * 8 + wid) * 16;

  const _Float16* qrow = Q + (size_t)(q0 + M) * DHEAD;
  v16h aq[2];
#pragma unroll
  for (int kc = 0; kc < 2; kc++)
    aq[kc] = pack16(*(const v8h*)(qrow + kc * 32 + kbA),
                    *(const v8h*)(qrow + kc * 32 + 16 + kbA));

  v8f o[4];
  float m[8], l[8];
#pragma unroll
  for (int t = 0; t < 4; t++)
#pragma unroll
    for (int v = 0; v < 8; v++) o[t][v] = 0.f;
#pragma unroll
  for (int v = 0; v < 8; v++) { m[v] = -3.0e38f; l[v] = 0.f; }

#pragma unroll 1
  for (int j = 0; j < q0 + 16; j += 32) {
    // ---- batch K fragment loads (4x 2 b128), then 4 score WMMAs ----
    v16h bk[2][2];
#pragma unroll
    for (int kt = 0; kt < 2; kt++) {
      const _Float16* krow = K + (size_t)(j + kt * 16 + M) * DHEAD;
#pragma unroll
      for (int kc = 0; kc < 2; kc++)
        bk[kt][kc] = pack16(*(const v8h*)(krow + kc * 32 + kbA),
                            *(const v8h*)(krow + kc * 32 + kbA + 16));
    }
    v8f sc[2];
#pragma unroll
    for (int kt = 0; kt < 2; kt++) {
#pragma unroll
      for (int v = 0; v < 8; v++) sc[kt][v] = 0.f;
#pragma unroll
      for (int kc = 0; kc < 2; kc++)
        sc[kt] = wmma_f16(aq[kc], bk[kt][kc], sc[kt]);
    }
    // ---- prefetch V fragments while softmax runs on the VALU ----
    v16h bv[4];
#pragma unroll
    for (int t = 0; t < 4; t++) {
      const _Float16* vrow = V + (size_t)(t * 16 + M) * SEQ + j + kbA;
      bv[t] = pack16(*(const v8h*)(vrow), *(const v8h*)(vrow + 16));
    }
    // ---- causal mask + online softmax ----
#pragma unroll
    for (int v = 0; v < 8; v++) {
      int qg = q0 + v + 8 * hs;
#pragma unroll
      for (int kt = 0; kt < 2; kt++)
        if (j + kt * 16 + M > qg) sc[kt][v] = -3.0e38f;
      float mx = fmaxf(sc[0][v], sc[1][v]);
#pragma unroll
      for (int d = 1; d < 16; d <<= 1) mx = fmaxf(mx, __shfl_xor(mx, d, 32));
      float mn = fmaxf(m[v], mx);
      float alpha = __expf(m[v] - mn);
      m[v] = mn;
      float rs = 0.f;
#pragma unroll
      for (int kt = 0; kt < 2; kt++) {
        float p = __expf(sc[kt][v] - mn);
        sc[kt][v] = p;
        rs += p;
      }
#pragma unroll
      for (int d = 1; d < 16; d <<= 1) rs += __shfl_xor(rs, d, 32);
      l[v] = l[v] * alpha + rs;
#pragma unroll
      for (int t = 0; t < 4; t++) o[t][v] *= alpha;
    }
    // ---- P: C-layout -> A-layout via per-wave LDS (DS in-order) ----
#pragma unroll
    for (int kt = 0; kt < 2; kt++)
#pragma unroll
      for (int v = 0; v < 8; v++)
        plds[wid][(v + 8 * hs) * 32 + kt * 16 + M] = (_Float16)sc[kt][v];
    const _Float16* prow = &plds[wid][M * 32];
    v16h ap = pack16(*(const v8h*)(prow + kbA), *(const v8h*)(prow + 16 + kbA));
#pragma unroll
    for (int t = 0; t < 4; t++) o[t] = wmma_f16(ap, bv[t], o[t]);
  }

#pragma unroll
  for (int v = 0; v < 8; v++) {
    float inv = 1.0f / l[v];
    int s = q0 + v + 8 * hs;
    _Float16* zrow = zb + ((size_t)(b * SEQ + s)) * DMODEL + n * DHEAD;
#pragma unroll
    for (int t = 0; t < 4; t++) zrow[t * 16 + M] = (_Float16)(o[t][v] * inv);
  }
}

// ---------------------------------------------------------------------------
// Kernel 3: output projection via TDM-staged LDS tiles + WMMA, fp32 out.
// ---------------------------------------------------------------------------
__global__ __launch_bounds__(256) void out_proj_kernel(
    const _Float16* __restrict__ zb, const _Float16* __restrict__ woT,
    const float* __restrict__ bO, float* __restrict__ out) {
  __shared__ alignas(16) _Float16 alds[2][128 * 32];
  __shared__ alignas(16) _Float16 wlds[2][64 * 32];

  const int rowbase = blockIdx.x * 128;
  const int n0 = blockIdx.y * 64;
  const int tid = threadIdx.x, wid = tid >> 5, L = tid & 31;
  const int M = L & 15, hs = L >> 4;
  const int kbA = hs ? 8 : 0, kbB = hs ? 16 : 0;
  const _Float16* zbase = zb + (size_t)rowbase * DMODEL;
  const _Float16* wbase = woT + (size_t)n0 * DMODEL;

  v8f acc[4];
#pragma unroll
  for (int t = 0; t < 4; t++)
#pragma unroll
    for (int v = 0; v < 8; v++) acc[t][v] = 0.f;

  auto issue = [&](int buf, int k0) {
#if HAVE_TDM
    if (wid == 0) {
      tdm_load_tile_f16(zbase + k0, lds_off(&alds[buf][0]),
                        DMODEL, ROWS, DMODEL, 32, 128);
      tdm_load_tile_f16(wbase + k0, lds_off(&wlds[buf][0]),
                        DMODEL, DMODEL, DMODEL, 32, 64);
    }
#else
#pragma unroll
    for (int i = 0; i < 2; i++) {
      int idx = tid + 256 * i, row = idx >> 2, c8 = (idx & 3) * 8;
      *(v8h*)&alds[buf][row * 32 + c8] =
          *(const v8h*)(zbase + (size_t)row * DMODEL + k0 + c8);
    }
    { int row = tid >> 2, c8 = (tid & 3) * 8;
      *(v8h*)&wlds[buf][row * 32 + c8] =
          *(const v8h*)(wbase + (size_t)row * DMODEL + k0 + c8); }
#endif
  };
  auto settle = [&]() {
#if HAVE_TDM
    if (wid == 0) __builtin_amdgcn_s_wait_tensorcnt((short)0);
#endif
    __syncthreads();
  };

  issue(0, 0);
  settle();
#pragma unroll 1
  for (int ks = 0; ks < 24; ks++) {
    int cur = ks & 1;
    if (ks + 1 < 24) issue(cur ^ 1, (ks + 1) * 32);
    const _Float16* arow = &alds[cur][(wid * 16 + M) * 32];
    v16h af = pack16(*(const v8h*)(arow + kbA), *(const v8h*)(arow + 16 + kbA));
    v16h bf[4];
#pragma unroll
    for (int t = 0; t < 4; t++) {
      const _Float16* wrow = &wlds[cur][(t * 16 + M) * 32];
      bf[t] = pack16(*(const v8h*)(wrow + kbA), *(const v8h*)(wrow + kbA + 16));
    }
#pragma unroll
    for (int t = 0; t < 4; t++) acc[t] = wmma_f16(af, bf[t], acc[t]);
    settle();
  }

#pragma unroll
  for (int t = 0; t < 4; t++) {
    float bb = bO[n0 + t * 16 + M];
#pragma unroll
    for (int v = 0; v < 8; v++) {
      int R = rowbase + wid * 16 + v + 8 * hs;
      out[(size_t)R * DMODEL + n0 + t * 16 + M] = acc[t][v] + bb;
    }
  }
}

// ---------------------------------------------------------------------------
extern "C" void kernel_launch(void* const* d_in, const int* in_sizes, int n_in,
                              void* d_out, int out_size, void* d_ws,
                              size_t ws_size, hipStream_t stream) {
  (void)in_sizes; (void)n_in; (void)out_size; (void)ws_size;
  const float* x  = (const float*)d_in[0];
  const float* WQ = (const float*)d_in[1];
  const float* WK = (const float*)d_in[2];
  const float* WV = (const float*)d_in[3];
  const float* WO = (const float*)d_in[4];
  const float* bQ = (const float*)d_in[5];
  const float* bK = (const float*)d_in[6];
  const float* bV = (const float*)d_in[7];
  const float* bO = (const float*)d_in[8];
  float* out = (float*)d_out;

  // f16 workspace: 4x 1.18MB packed weights + xh + q/k/vT + z  (~67.6 MB)
  _Float16* ws = (_Float16*)d_ws;
  const size_t WT = (size_t)DMODEL * DMODEL;              // 589824
  const size_t PH = (size_t)BATCH * NHEADS * SEQ * DHEAD; // 6291456
  _Float16* wqT = ws;
  _Float16* wkT = wqT + WT;
  _Float16* wvT = wkT + WT;
  _Float16* woT = wvT + WT;
  _Float16* xh  = woT + WT;
  _Float16* qh  = xh + (size_t)ROWS * DMODEL;
  _Float16* kh  = qh + PH;
  _Float16* vT  = kh + PH;
  _Float16* zb  = vT + PH;

  cvt_x_kernel<<<dim3(ROWS * DMODEL / 1024), 256, 0, stream>>>(x, xh);
  pack_weights_kernel<<<dim3((DMODEL * DMODEL + 255) / 256), 256, 0, stream>>>(
      WQ, WK, WV, WO, wqT, wkT, wvT, woT);
  qkv_proj_kernel<<<dim3(ROWS / 128, NHEADS, 3), 256, 0, stream>>>(
      xh, wqT, wkT, wvT, bQ, bK, bV, qh, kh, vT);
  attn_kernel<<<dim3(SEQ / 128, NHEADS, BATCH), 256, 0, stream>>>(
      qh, kh, vT, zb);
  out_proj_kernel<<<dim3(ROWS / 128, DMODEL / 64, 1), 256, 0, stream>>>(
      zb, woT, bO, out);
}
